// TPMSAttention_15814069584145
// MI455X (gfx1250) — matrix operand order repaired and checked by reading, not compiled, not run on hardware
//
#include <hip/hip_runtime.h>
#include <hip/hip_bf16.h>

typedef __attribute__((ext_vector_type(16))) _Float16 v16h;
typedef __attribute__((ext_vector_type(8)))  float    v8f;

#define NH   16
#define HD   64
#define LQ   2048
#define HDIM 1024
#define BB   2
#define TWO_PI 6.283185307179586f
#define LOG2E 1.4426950408889634f
#define LN2_X03 0.20794415416798357f   // 0.3 * ln(2)

// ---------------------------------------------------------------------------
// Async global->LDS (CDNA5 GLOBAL_LOAD_ASYNC_TO_LDS_B128, ASYNCcnt-tracked).
// Signature per hipcc diagnostic: param0 = <4 x i32> AS1*, so pass v4i pointers.
// Guarded: falls back to VGPR-staged copies if the builtin isn't declared.
// ---------------------------------------------------------------------------
#if defined(__has_builtin)
#if __has_builtin(__builtin_amdgcn_global_load_async_to_lds_b128)
#define HAVE_ASYNC_LDS 1
#endif
#endif

typedef int v4i_vs __attribute__((vector_size(16)));
typedef __attribute__((address_space(1))) v4i_vs* gptr128_t;
typedef __attribute__((address_space(3))) v4i_vs* lptr128_t;

__device__ __forceinline__ void cp16_lds(void* lds, const void* g) {
#ifdef HAVE_ASYNC_LDS
    void* gg = const_cast<void*>(g);
    __builtin_amdgcn_global_load_async_to_lds_b128(
        (gptr128_t)gg, (lptr128_t)lds, 0, 0);
#else
    *reinterpret_cast<uint4*>(lds) = *reinterpret_cast<const uint4*>(g);
#endif
}

__device__ __forceinline__ void async_wait_all() {
#ifdef HAVE_ASYNC_LDS
#if __has_builtin(__builtin_amdgcn_s_wait_asynccnt)
    __builtin_amdgcn_s_wait_asynccnt(0);
#else
    asm volatile("s_wait_asynccnt 0" ::: "memory");
#endif
#endif
}

// Raw hardware ops: single-instruction rcp / exp2 / log2 (args never denormal here).
__device__ __forceinline__ float fast_rcp(float x)  { return __builtin_amdgcn_rcpf(x); }
__device__ __forceinline__ float fast_exp2(float x) { return __builtin_amdgcn_exp2f(x); }
__device__ __forceinline__ float fast_log2(float x) { return __builtin_amdgcn_logf(x); }

// ---------------------------------------------------------------------------
// f32 -> f16 conversion (grid-stride)
// ---------------------------------------------------------------------------
__global__ __launch_bounds__(256) void cvt_f32_f16(const float* __restrict__ in,
                                                   _Float16* __restrict__ out, int n) {
    int i = blockIdx.x * blockDim.x + threadIdx.x;
    int stride = gridDim.x * blockDim.x;
    for (; i < n; i += stride) out[i] = (_Float16)in[i];
}

// ---------------------------------------------------------------------------
// Tiled WMMA GEMM: C[M,N] = A[M,K](f16) * B[K,N](f16) + bias[N]
// Block tile 128x128, 256 threads = 8 wave32 waves, each wave: 16 rows x 128 cols.
// Double-buffered LDS; async DMA of tile k+1 overlaps WMMA on tile k.
// ---------------------------------------------------------------------------
template <bool OUT_F16>
__global__ __launch_bounds__(256) void gemm_wmma_kernel(
    const _Float16* __restrict__ A, const _Float16* __restrict__ Bm,
    const float* __restrict__ bias, void* __restrict__ outv,
    int M, int N, int K)
{
    __shared__ _Float16 As[2][128][40];    // row pitch 80B (16B-aligned)
    __shared__ _Float16 Bs[2][32][136];    // row pitch 272B (16B-aligned)

    const int tid  = threadIdx.x;
    const int lane = tid & 31;
    const int wv   = tid >> 5;
    const int n0   = blockIdx.x * 128;
    const int m0   = blockIdx.y * 128;

    const int lm  = lane & 15;
    const int alo = (lane < 16) ? 0 : 8;    // half the A-frag K offset; used as alo<<1 (contiguous K)
    const int bko = (lane < 16) ? 0 : 16;   // B-frag contraction offset

    v8f acc[8];
#pragma unroll
    for (int t = 0; t < 8; ++t)
#pragma unroll
        for (int j = 0; j < 8; ++j) acc[t][j] = 0.0f;

    const int arow = tid >> 1, acb = (tid & 1) * 16;   // A tile: 128x32 halves
    const int brow = tid >> 3, bcb = (tid & 7) * 16;   // B tile: 32x128 halves

    auto issue_tile = [&](int buf, int k0) {
        const _Float16* ag = A  + (size_t)(m0 + arow) * K + k0 + acb;
        const _Float16* bg = Bm + (size_t)(k0 + brow) * N + n0 + bcb;
        cp16_lds(&As[buf][arow][acb],     ag);
        cp16_lds(&As[buf][arow][acb + 8], ag + 8);
        cp16_lds(&Bs[buf][brow][bcb],     bg);
        cp16_lds(&Bs[buf][brow][bcb + 8], bg + 8);
    };

    const int NT = K >> 5;
    issue_tile(0, 0);
    for (int kt = 0; kt < NT; ++kt) {
        const int cur = kt & 1;
        async_wait_all();
        __syncthreads();                       // buf `cur` filled; all waves done reading `cur^1`
        if (kt + 1 < NT) issue_tile(cur ^ 1, (kt + 1) << 5);

        // A fragment: lane = M, elems 0..15 -> K = 16*(lane>=16) + e (contiguous)
        v16h af;
#pragma unroll
        for (int i = 0; i < 8; ++i) af[i]     = As[cur][wv * 16 + lm][(alo << 1) + i];
#pragma unroll
        for (int i = 0; i < 8; ++i) af[8 + i] = As[cur][wv * 16 + lm][(alo << 1) + 8 + i];

#pragma unroll
        for (int nt = 0; nt < 8; ++nt) {
            v16h bf;   // lane = N, elems = contiguous contraction chunk
#pragma unroll
            for (int i = 0; i < 16; ++i) bf[i] = Bs[cur][bko + i][nt * 16 + lm];
            acc[nt] = __builtin_amdgcn_wmma_f32_16x16x32_f16(
                false, af, false, bf, (short)0, acc[nt], false, false);
        }
    }

    // Epilogue: C layout -> M = r + 8*(lane>=16), N = lane&15
#pragma unroll
    for (int nt = 0; nt < 8; ++nt) {
#pragma unroll
        for (int r = 0; r < 8; ++r) {
            const int mm = m0 + wv * 16 + r + ((lane < 16) ? 0 : 8);
            const int nn = n0 + nt * 16 + lm;
            const float v = acc[nt][r] + bias[nn];
            if (OUT_F16) ((_Float16*)outv)[(size_t)mm * N + nn] = (_Float16)v;
            else         ((float*)outv)[(size_t)mm * N + nn]    = v;
        }
    }
}

// ---------------------------------------------------------------------------
// TPMS flash-attention. Grid: (L/128, NH, B), 256 threads (8 waves).
// Each wave owns a 16-row Q strip; double-buffered 32-wide KV chunks in LDS.
// Mask computed on the fly with transcendentals (co-execute with XDL WMMA).
// ---------------------------------------------------------------------------
__global__ __launch_bounds__(256) void tpms_attn_kernel(
    const _Float16* __restrict__ Qh, const _Float16* __restrict__ Kh,
    const _Float16* __restrict__ Vh, const float* __restrict__ phase,
    const float* __restrict__ thr_p, const float* __restrict__ temp_p,
    _Float16* __restrict__ Oh)
{
    __shared__ _Float16 Ks[2][32][72];   // row pitch 144B (16B-aligned)
    __shared__ _Float16 Vs[2][32][72];
    __shared__ _Float16 Pb[8][16][36];   // per-wave P bounce: C-layout -> A-layout

    const int tid = threadIdx.x, lane = tid & 31, wv = tid >> 5;
    const int b = blockIdx.z, h = blockIdx.y;
    const int q0 = blockIdx.x * 128 + wv * 16;
    const int lm  = lane & 15;
    const int mof = (lane < 16) ? 0 : 8;
    const int alo = mof;
    const int bko = (lane < 16) ? 0 : 16;

    const float ph0 = phase[h * 3 + 0], ph1 = phase[h * 3 + 1], ph2 = phase[h * 3 + 2];
    const float thr  = thr_p[0];
    // sigmoid(v/T) = 1/(1 + 2^(v * negInvT_log2e))
    const float negInvT_log2e = -LOG2E * fast_rcp(temp_p[0] + 1e-6f);
    const float posScale = TWO_PI / (float)LQ;

    // Q fragments: hd=64 -> two K=32 A-frags, loaded once
    v16h qf[2];
    const _Float16* qbase = Qh + (size_t)(b * LQ + q0 + lm) * HDIM + h * HD;
#pragma unroll
    for (int c = 0; c < 2; ++c) {
#pragma unroll
        for (int i = 0; i < 8; ++i) qf[c][i]     = qbase[c * 32 + (alo << 1) + i];
#pragma unroll
        for (int i = 0; i < 8; ++i) qf[c][8 + i] = qbase[c * 32 + (alo << 1) + 8 + i];
    }

    // Hoisted per-row trig (q fixed for the whole kv loop)
    float posq[8], sq[8], cq[8];
#pragma unroll
    for (int r = 0; r < 8; ++r) {
        posq[r] = (float)(q0 + r + mof) * posScale;
        sq[r] = __sinf(posq[r] + ph0);
        cq[r] = __cosf(posq[r] + ph0);
    }

    float mrun[8], lrun[8];
    v8f o[4];
#pragma unroll
    for (int r = 0; r < 8; ++r) { mrun[r] = -1e30f; lrun[r] = 0.0f; }
#pragma unroll
    for (int dt = 0; dt < 4; ++dt)
#pragma unroll
        for (int r = 0; r < 8; ++r) o[dt][r] = 0.0f;

    const int kvrow = tid >> 3, kvcb = (tid & 7) * 8;   // 32 rows x 64 cols, 16B/thread

    auto issue_kv = [&](int buf, int k0) {
        const _Float16* kg = Kh + (size_t)(b * LQ + k0 + kvrow) * HDIM + h * HD + kvcb;
        const _Float16* vg = Vh + (size_t)(b * LQ + k0 + kvrow) * HDIM + h * HD + kvcb;
        cp16_lds(&Ks[buf][kvrow][kvcb], kg);
        cp16_lds(&Vs[buf][kvrow][kvcb], vg);
    };

    issue_kv(0, 0);
    for (int k0 = 0; k0 < LQ; k0 += 32) {
        const int cur = (k0 >> 5) & 1;
        async_wait_all();
        __syncthreads();                        // KV chunk `cur` in LDS; prior readers done
        if (k0 + 32 < LQ) issue_kv(cur ^ 1, k0 + 32);

        // S = Q K^T for two 16-col tiles; fold in gyroid mask
        float sc[2][8];
#pragma unroll
        for (int nt = 0; nt < 2; ++nt) {
            v8f s;
#pragma unroll
            for (int j = 0; j < 8; ++j) s[j] = 0.0f;
#pragma unroll
            for (int c = 0; c < 2; ++c) {
                v16h bf;   // B[d][n] = K[n][d]: lane n reads contiguous d-chunk of K row
#pragma unroll
                for (int i = 0; i < 16; ++i) bf[i] = Ks[cur][nt * 16 + lm][c * 32 + bko + i];
                s = __builtin_amdgcn_wmma_f32_16x16x32_f16(
                    false, qf[c], false, bf, (short)0, s, false, false);
            }
            const float posk = (float)(k0 + nt * 16 + lm) * posScale;
            const float spk = __sinf(posk + ph1);
            const float cpk = __cosf(posk + ph1);
#pragma unroll
            for (int r = 0; r < 8; ++r) {
                const float pz  = (posq[r] + posk) * 0.5f + ph2;
                const float val = sq[r] * cpk + spk * __cosf(pz) + __sinf(pz) * cq[r] - thr;
                // sigmoid via raw v_exp/v_rcp; 0.3*ln(sig+eps) via raw v_log (log2) * 0.3*ln2.
                const float sig = fast_rcp(1.0f + fast_exp2(val * negInvT_log2e));
                sc[nt][r] = s[r] * 0.125f + LN2_X03 * fast_log2(sig + 1e-6f);
            }
        }

        // Online softmax: 16-lane butterfly reductions (wave32 halves)
#pragma unroll
        for (int r = 0; r < 8; ++r) {
            float t = fmaxf(sc[0][r], sc[1][r]);
#pragma unroll
            for (int off = 1; off < 16; off <<= 1) t = fmaxf(t, __shfl_xor(t, off, 32));
            const float mnew  = fmaxf(mrun[r], t);
            const float alpha = fast_exp2((mrun[r] - mnew) * LOG2E);
            const float p0 = fast_exp2((sc[0][r] - mnew) * LOG2E);
            const float p1 = fast_exp2((sc[1][r] - mnew) * LOG2E);
            float rs = p0 + p1;
#pragma unroll
            for (int off = 1; off < 16; off <<= 1) rs += __shfl_xor(rs, off, 32);
            lrun[r] = lrun[r] * alpha + rs;
            mrun[r] = mnew;
#pragma unroll
            for (int dt = 0; dt < 4; ++dt) o[dt][r] *= alpha;
            Pb[wv][r + mof][lm]      = (_Float16)p0;   // kv col nt=0
            Pb[wv][r + mof][16 + lm] = (_Float16)p1;   // kv col nt=1
        }
        __syncthreads();   // Pb visible across lanes before A-frag reads

        // P (16x32) as A-fragment, V chunk as B-fragments: O += P*V
        v16h pf;
#pragma unroll
        for (int i = 0; i < 8; ++i) pf[i]     = Pb[wv][lm][(alo << 1) + i];
#pragma unroll
        for (int i = 0; i < 8; ++i) pf[8 + i] = Pb[wv][lm][(alo << 1) + 8 + i];
#pragma unroll
        for (int dt = 0; dt < 4; ++dt) {
            v16h bf;
#pragma unroll
            for (int i = 0; i < 16; ++i) bf[i] = Vs[cur][bko + i][dt * 16 + lm];
            o[dt] = __builtin_amdgcn_wmma_f32_16x16x32_f16(
                false, pf, false, bf, (short)0, o[dt], false, false);
        }
    }

    // Normalize (single v_rcp_f32 per row) + store f16 for the Wo GEMM
    float il[8];
#pragma unroll
    for (int r = 0; r < 8; ++r) il[r] = fast_rcp(lrun[r]);
#pragma unroll
    for (int dt = 0; dt < 4; ++dt)
#pragma unroll
        for (int r = 0; r < 8; ++r) {
            const float v = o[dt][r] * il[r];
            Oh[(size_t)(b * LQ + q0 + r + mof) * HDIM + h * HD + dt * 16 + lm] = (_Float16)v;
        }
}

// ---------------------------------------------------------------------------
// Host launch
// ---------------------------------------------------------------------------
extern "C" void kernel_launch(void* const* d_in, const int* in_sizes, int n_in,
                              void* d_out, int out_size, void* d_ws, size_t ws_size,
                              hipStream_t stream) {
    const float* x     = (const float*)d_in[0];
    const float* Wq    = (const float*)d_in[1];
    const float* bq    = (const float*)d_in[2];
    const float* Wk    = (const float*)d_in[3];
    const float* bk    = (const float*)d_in[4];
    const float* Wv    = (const float*)d_in[5];
    const float* bv    = (const float*)d_in[6];
    const float* Wo    = (const float*)d_in[7];
    const float* bo    = (const float*)d_in[8];
    const float* phase = (const float*)d_in[9];
    const float* thr   = (const float*)d_in[10];
    const float* temp  = (const float*)d_in[11];

    const int XE = BB * LQ * HDIM;      // 4,194,304
    const int WE = HDIM * HDIM;         // 1,048,576

    char* ws = (char*)d_ws;
    size_t off = 0;
    _Float16* xh  = (_Float16*)(ws + off); off += (size_t)XE * 2;
    _Float16* wqh = (_Float16*)(ws + off); off += (size_t)WE * 2;
    _Float16* wkh = (_Float16*)(ws + off); off += (size_t)WE * 2;
    _Float16* wvh = (_Float16*)(ws + off); off += (size_t)WE * 2;
    _Float16* woh = (_Float16*)(ws + off); off += (size_t)WE * 2;
    _Float16* qh  = (_Float16*)(ws + off); off += (size_t)XE * 2;
    _Float16* kh  = (_Float16*)(ws + off); off += (size_t)XE * 2;
    _Float16* vh  = (_Float16*)(ws + off); off += (size_t)XE * 2;
    _Float16* ah  = (_Float16*)(ws + off); off += (size_t)XE * 2;   // ~48 MB total
    (void)ws_size; (void)in_sizes; (void)n_in; (void)out_size;

    cvt_f32_f16<<<2048, 256, 0, stream>>>(x,  xh,  XE);
    cvt_f32_f16<<<1024, 256, 0, stream>>>(Wq, wqh, WE);
    cvt_f32_f16<<<1024, 256, 0, stream>>>(Wk, wkh, WE);
    cvt_f32_f16<<<1024, 256, 0, stream>>>(Wv, wvh, WE);
    cvt_f32_f16<<<1024, 256, 0, stream>>>(Wo, woh, WE);

    const dim3 gg(HDIM / 128, (BB * LQ) / 128);   // (8, 32)
    gemm_wmma_kernel<true ><<<gg, 256, 0, stream>>>(xh, wqh, bq, qh, BB * LQ, HDIM, HDIM);
    gemm_wmma_kernel<true ><<<gg, 256, 0, stream>>>(xh, wkh, bk, kh, BB * LQ, HDIM, HDIM);
    gemm_wmma_kernel<true ><<<gg, 256, 0, stream>>>(xh, wvh, bv, vh, BB * LQ, HDIM, HDIM);

    tpms_attn_kernel<<<dim3(LQ / 128, NH, BB), 256, 0, stream>>>(
        qh, kh, vh, phase, thr, temp, ah);

    gemm_wmma_kernel<false><<<gg, 256, 0, stream>>>(ah, woh, bo, d_out, BB * LQ, HDIM, HDIM);
}
